// DomainGCN_62045097558307
// MI455X (gfx1250) — matrix operand order repaired and checked by reading, not compiled, not run on hardware
//
#include <hip/hip_runtime.h>

// ---------------------------------------------------------------------------
// Types for CDNA5 WMMA (wave32, 16x16x32 bf16 -> f32 accum)
// ---------------------------------------------------------------------------
typedef __attribute__((ext_vector_type(16))) __bf16         v16bf;
typedef __attribute__((ext_vector_type(8)))  float          v8f;
typedef __attribute__((ext_vector_type(4)))  unsigned short us4;
typedef __attribute__((ext_vector_type(8)))  unsigned short us8;
typedef __attribute__((ext_vector_type(16))) unsigned short us16;

__device__ __forceinline__ unsigned short f32_to_bf16(float f) {
  unsigned u = __builtin_bit_cast(unsigned, f);
  u += 0x7FFFu + ((u >> 16) & 1u);   // round-to-nearest-even
  return (unsigned short)(u >> 16);
}

// ---------------------------------------------------------------------------
// Small helper kernels: degree / rsqrt / edge coefficients
// ---------------------------------------------------------------------------
__global__ void fill_k(float* __restrict__ p, float v, int n) {
  int i = blockIdx.x * blockDim.x + threadIdx.x;
  if (i < n) p[i] = v;
}

__global__ void deg_k(const int* __restrict__ dst, float* __restrict__ deg, int E) {
  int e = blockIdx.x * blockDim.x + threadIdx.x;
  if (e < E) atomicAdd(&deg[dst[e]], 1.0f);
}

__global__ void rsqrt_k(float* __restrict__ p, int n) {
  int i = blockIdx.x * blockDim.x + threadIdx.x;
  if (i < n) p[i] = rsqrtf(p[i]);
}

__global__ void coef_k(const int* __restrict__ src, const int* __restrict__ dst,
                       const float* __restrict__ dis, float* __restrict__ coef, int E) {
  int e = blockIdx.x * blockDim.x + threadIdx.x;
  if (e < E) coef[e] = dis[src[e]] * dis[dst[e]];
}

// ---------------------------------------------------------------------------
// f32 -> bf16 conversion (optionally fused ReLU), 4 elems/thread
// ---------------------------------------------------------------------------
__global__ void cvt_bf16_k(const float* __restrict__ in, unsigned short* __restrict__ out,
                           long long n4, int relu) {
  long long i = (long long)blockIdx.x * blockDim.x + threadIdx.x;
  if (i >= n4) return;
  float4 v = reinterpret_cast<const float4*>(in)[i];
  if (relu) {
    v.x = fmaxf(v.x, 0.f); v.y = fmaxf(v.y, 0.f);
    v.z = fmaxf(v.z, 0.f); v.w = fmaxf(v.w, 0.f);
  }
  us4 o;
  o.s0 = f32_to_bf16(v.x); o.s1 = f32_to_bf16(v.y);
  o.s2 = f32_to_bf16(v.z); o.s3 = f32_to_bf16(v.w);
  reinterpret_cast<us4*>(out)[i] = o;
}

// Transpose+convert weights: Wt[n][k] = bf16(W[k][n]); rows n >= Ncols zero-padded.
__global__ void cvt_wt_k(const float* __restrict__ W, unsigned short* __restrict__ Wt,
                         int K, int Ncols, int Npad) {
  int i = blockIdx.x * blockDim.x + threadIdx.x;
  if (i >= Npad * K) return;
  int n = i / K, k = i - n * K;
  float v = (n < Ncols) ? W[(size_t)k * Ncols + n] : 0.0f;
  Wt[i] = f32_to_bf16(v);
}

// ---------------------------------------------------------------------------
// WMMA GEMM: C[M x Ncols] = A[M x 512](bf16) * Wt^T, Wt is [Npad x 512] bf16.
// grid.x = M/16, grid.y = ceil(Ncols/64); block = 128 (4 waves, 16x16 tile each)
// ---------------------------------------------------------------------------
__launch_bounds__(128)
__global__ void wmma_gemm_bf16(const unsigned short* __restrict__ A,
                               const unsigned short* __restrict__ Bt,
                               float* __restrict__ C,
                               int Ncols, int ldc) {
  constexpr int K = 512;
  __shared__ __align__(16) unsigned short As[16][K];   // 16 KB A tile

  const int m0  = blockIdx.x * 16;
  const int tid = threadIdx.x;

  { // cooperative, fully coalesced A-tile stage: 1024 x 16B
    const us8* sv = reinterpret_cast<const us8*>(A + (size_t)m0 * K);
    us8* dv = reinterpret_cast<us8*>(&As[0][0]);
    #pragma unroll
    for (int i = 0; i < (16 * K / 8) / 128; ++i)
      dv[tid + i * 128] = sv[tid + i * 128];
  }
  __syncthreads();

  const int wave = tid >> 5;
  const int lane = tid & 31;
  const int n0   = (blockIdx.y * 4 + wave) * 16;
  if (n0 >= Ncols) return;

  const int nc = lane & 15;                 // A row within tile == output col within tile
  const int ah = (lane < 16) ? 0 : 8;       // A K-half select (ISA 16-bit A layout)
  const int bh = (lane < 16) ? 0 : 8;       // B K-half select (same K order as A)
  const unsigned short* Brow = Bt + (size_t)(n0 + nc) * K;

  v8f acc = {0.f, 0.f, 0.f, 0.f, 0.f, 0.f, 0.f, 0.f};

  #pragma unroll 4
  for (int kb = 0; kb < K; kb += 32) {
    us8 a_lo = *reinterpret_cast<const us8*>(&As[nc][kb + ah]);
    us8 a_hi = *reinterpret_cast<const us8*>(&As[nc][kb + 16 + ah]);
    us8 b_lo = *reinterpret_cast<const us8*>(Brow + kb + bh);
    us8 b_hi = *reinterpret_cast<const us8*>(Brow + kb + 16 + bh);
    us16 au = __builtin_shufflevector(a_lo, a_hi, 0,1,2,3,4,5,6,7,8,9,10,11,12,13,14,15);
    us16 bu = __builtin_shufflevector(b_lo, b_hi, 0,1,2,3,4,5,6,7,8,9,10,11,12,13,14,15);
    v16bf av = __builtin_bit_cast(v16bf, au);
    v16bf bv = __builtin_bit_cast(v16bf, bu);
    acc = __builtin_amdgcn_wmma_f32_16x16x32_bf16(
        /*neg_a=*/false, av, /*neg_b=*/false, bv,
        /*c_mod=*/(short)0, acc, /*reuse_a=*/false, /*reuse_b=*/false);
  }

  const int rbase = (lane < 16) ? 0 : 8;    // C/D layout: vgpr r -> row r (+8 upper lanes)
  float* Cp = C + (size_t)(m0 + rbase) * ldc + n0 + nc;
  #pragma unroll
  for (int r = 0; r < 8; ++r)
    Cp[(size_t)r * ldc] = acc[r];
}

// ---------------------------------------------------------------------------
// Self-loop + bias init: agg[i][d] = hw[i][d]*dis[i]^2 + b[d]   (4 elems/thread)
// ---------------------------------------------------------------------------
__global__ void self_init_k(const float* __restrict__ hw, const float* __restrict__ dis,
                            const float* __restrict__ bias, float* __restrict__ agg,
                            int per /* Dpad/4 */, int Dvalid, long long total4) {
  long long i = (long long)blockIdx.x * blockDim.x + threadIdx.x;
  if (i >= total4) return;
  int node = (int)(i / per);
  int d0   = (int)(i - (long long)node * per) * 4;
  float s = dis[node]; s = s * s;
  float4 v = reinterpret_cast<const float4*>(hw)[i];
  float b0 = (d0 + 0 < Dvalid) ? bias[d0 + 0] : 0.f;
  float b1 = (d0 + 1 < Dvalid) ? bias[d0 + 1] : 0.f;
  float b2 = (d0 + 2 < Dvalid) ? bias[d0 + 2] : 0.f;
  float b3 = (d0 + 3 < Dvalid) ? bias[d0 + 3] : 0.f;
  float4 o; o.x = v.x * s + b0; o.y = v.y * s + b1; o.z = v.z * s + b2; o.w = v.w * s + b3;
  reinterpret_cast<float4*>(agg)[i] = o;
}

// ---------------------------------------------------------------------------
// Edge scatter, D=512: one wave32 per edge, each lane owns 16 channels.
// Traffic per edge = 2KB read + 2KB atomic-add == roofline minimum.
// ---------------------------------------------------------------------------
__global__ void scatter512_k(const int* __restrict__ src, const int* __restrict__ dst,
                             const float* __restrict__ coef,
                             const float* __restrict__ hw, float* __restrict__ agg, int E) {
  int t = blockIdx.x * blockDim.x + threadIdx.x;
  int e = t >> 5;
  int lane = t & 31;
  if (e >= E) return;
  int s = src[e], d = dst[e];
  float c = coef[e];
  const float4* hs = reinterpret_cast<const float4*>(hw + (size_t)s * 512 + lane * 16);
  float* ad = agg + (size_t)d * 512 + lane * 16;
  #pragma unroll
  for (int i = 0; i < 4; ++i) {
    float4 v = hs[i];
    atomicAdd(ad + i * 4 + 0, v.x * c);
    atomicAdd(ad + i * 4 + 1, v.y * c);
    atomicAdd(ad + i * 4 + 2, v.z * c);
    atomicAdd(ad + i * 4 + 3, v.w * c);
  }
}

// Edge scatter for layer 4 (10 valid channels, padded stride 16): 1 thread/edge
__global__ void scatter16_k(const int* __restrict__ src, const int* __restrict__ dst,
                            const float* __restrict__ coef,
                            const float* __restrict__ hw4, float* __restrict__ agg4, int E) {
  int e = blockIdx.x * blockDim.x + threadIdx.x;
  if (e >= E) return;
  int s = src[e], d = dst[e];
  float c = coef[e];
  const float* h = hw4 + (size_t)s * 16;
  float* a = agg4 + (size_t)d * 16;
  #pragma unroll
  for (int i = 0; i < 10; ++i)
    atomicAdd(a + i, h[i] * c);
}

// ---------------------------------------------------------------------------
// MLP head: relu(in[10]) -> 64 -> 64 -> 20, one thread per node (~6K FMA/node)
// ---------------------------------------------------------------------------
__global__ void mlp_k(const float* __restrict__ agg4,
                      const float* __restrict__ M1, const float* __restrict__ mb1,
                      const float* __restrict__ M2, const float* __restrict__ mb2,
                      const float* __restrict__ M3, const float* __restrict__ mb3,
                      float* __restrict__ out, int N) {
  int n = blockIdx.x * blockDim.x + threadIdx.x;
  if (n >= N) return;
  float in[10];
  #pragma unroll
  for (int k = 0; k < 10; ++k) in[k] = fmaxf(agg4[(size_t)n * 16 + k], 0.f);
  float h1[64];
  #pragma unroll
  for (int j = 0; j < 64; ++j) {
    float a = mb1[j];
    #pragma unroll
    for (int k = 0; k < 10; ++k) a += in[k] * M1[k * 64 + j];
    h1[j] = fmaxf(a, 0.f);
  }
  float h2[64];
  #pragma unroll
  for (int j = 0; j < 64; ++j) {
    float a = mb2[j];
    #pragma unroll
    for (int k = 0; k < 64; ++k) a += h1[k] * M2[k * 64 + j];
    h2[j] = fmaxf(a, 0.f);
  }
  #pragma unroll
  for (int j = 0; j < 20; ++j) {
    float a = mb3[j];
    #pragma unroll
    for (int k = 0; k < 64; ++k) a += h2[k] * M3[k * 20 + j];
    out[(size_t)n * 20 + j] = a;
  }
}

// ---------------------------------------------------------------------------
// Host orchestration
// ---------------------------------------------------------------------------
extern "C" void kernel_launch(void* const* d_in, const int* in_sizes, int n_in,
                              void* d_out, int out_size, void* d_ws, size_t ws_size,
                              hipStream_t stream) {
  const int DIN = 512, DH = 512;
  const int N = in_sizes[0] / DIN;       // 50000
  const int E = in_sizes[1] / 2;         // 800000

  const float* x   = (const float*)d_in[0];
  const int*   ei  = (const int*)d_in[1];
  const int*   src = ei;
  const int*   dst = ei + E;
  const float* W1 = (const float*)d_in[2];  const float* b1 = (const float*)d_in[3];
  const float* W2 = (const float*)d_in[4];  const float* b2 = (const float*)d_in[5];
  const float* W3 = (const float*)d_in[6];  const float* b3 = (const float*)d_in[7];
  const float* W4 = (const float*)d_in[8];  const float* b4 = (const float*)d_in[9];
  const float* M1 = (const float*)d_in[10]; const float* mb1 = (const float*)d_in[11];
  const float* M2 = (const float*)d_in[12]; const float* mb2 = (const float*)d_in[13];
  const float* M3 = (const float*)d_in[14]; const float* mb3 = (const float*)d_in[15];
  float* out = (float*)d_out;

  // Carve scratch from d_ws (256B aligned slices)
  char* ws = (char*)d_ws;
  size_t off = 0;
  auto carve = [&](size_t bytes) -> void* {
    void* p = ws + off;
    off += (bytes + 255) & ~(size_t)255;
    return p;
  };
  float*          dis  = (float*)carve((size_t)N * 4);
  float*          coef = (float*)carve((size_t)E * 4);
  unsigned short* hb   = (unsigned short*)carve((size_t)N * DH * 2);
  unsigned short* Wt   = (unsigned short*)carve((size_t)DH * DH * 2);
  float*          hw   = (float*)carve((size_t)N * DH * 4);
  float*          agg  = (float*)carve((size_t)N * DH * 4);
  float*          agg4 = (float*)carve((size_t)N * 16 * 4);

  const int T = 256;
  auto blk = [&](long long n) { return (unsigned)((n + T - 1) / T); };

  // 1) degrees (self-loop via fill 1.0), dis = rsqrt(deg), per-edge coefficients
  fill_k <<<blk(N), T, 0, stream>>>(dis, 1.0f, N);
  deg_k  <<<blk(E), T, 0, stream>>>(dst, dis, E);
  rsqrt_k<<<blk(N), T, 0, stream>>>(dis, N);
  coef_k <<<blk(E), T, 0, stream>>>(src, dst, dis, coef, E);

  const long long n4_512 = (long long)N * DH / 4;
  const dim3 gemm_grid(N / 16, DH / 64);           // 3125 x 8
  const unsigned scat_grid = blk((long long)E * 32);

  // 2) GCN layers 1..3 (512 -> 512)
  const float* actW[3] = {W1, W2, W3};
  const float* actB[3] = {b1, b2, b3};
  for (int L = 0; L < 3; ++L) {
    const float* actin = (L == 0) ? x : agg;
    int relu = (L == 0) ? 0 : 1;                   // ReLU of previous layer fused here
    cvt_bf16_k<<<blk(n4_512), T, 0, stream>>>(actin, hb, n4_512, relu);
    cvt_wt_k  <<<blk(DH * DH), T, 0, stream>>>(actW[L], Wt, DH, DH, DH);
    wmma_gemm_bf16<<<gemm_grid, 128, 0, stream>>>(hb, Wt, hw, DH, DH);
    self_init_k<<<blk(n4_512), T, 0, stream>>>(hw, dis, actB[L], agg, DH / 4, DH, n4_512);
    scatter512_k<<<scat_grid, T, 0, stream>>>(src, dst, coef, hw, agg, E);
  }

  // 3) GCN layer 4 (512 -> 10, N padded to 16 for WMMA)
  cvt_bf16_k<<<blk(n4_512), T, 0, stream>>>(agg, hb, n4_512, 1);
  cvt_wt_k  <<<blk(16 * DH), T, 0, stream>>>(W4, Wt, DH, 10, 16);
  wmma_gemm_bf16<<<dim3(N / 16, 1), 128, 0, stream>>>(hb, Wt, hw, 16, 16);
  const long long n4_16 = (long long)N * 16 / 4;
  self_init_k<<<blk(n4_16), T, 0, stream>>>(hw, dis, b4, agg4, 4, 10, n4_16);
  scatter16_k<<<blk(E), T, 0, stream>>>(src, dst, coef, hw, agg4, E);

  // 4) MLP head -> d_out [N x 20] f32
  mlp_k<<<blk(N), T, 0, stream>>>(agg4, M1, mb1, M2, mb2, M3, mb3, out, N);
}
